// MHCA_55843164783199
// MI455X (gfx1250) — matrix operand order repaired and checked by reading, not compiled, not run on hardware
//
#include <hip/hip_runtime.h>
#include <cstdint>
#include <cstddef>

#define DEV __device__ __forceinline__

typedef __attribute__((ext_vector_type(16))) __bf16 bf16x16;
typedef __attribute__((ext_vector_type(8)))  float  f32x8;

static constexpr int Dc   = 192;
static constexpr int D2c  = 384;
static constexpr int Hc   = 48;
static constexpr int Wc   = 48;
static constexpr int BSc  = 2;
static constexpr int NYc  = Hc * Wc;      // 2304 tokens per image
static constexpr int NTOK = BSc * NYc;    // 4608 tokens total
static constexpr int NHc  = 8;
static constexpr int HDQK = D2c / NHc;    // 48
static constexpr int HDV  = Dc / NHc;     // 24
static constexpr int PW   = Wc + 2;       // 50 (zero halo)
static constexpr int PTOK = PW * PW;      // 2500 padded slots per image

union FragU { uint4 q[2]; bf16x16 v; };

DEV uint4 ld16(const unsigned short* p) { return *(const uint4*)p; }

DEV unsigned short f2bf(float f) {            // fp32 -> bf16 RNE
  unsigned u = __float_as_uint(f);
  u += 0x7FFFu + ((u >> 16) & 1u);
  return (unsigned short)(u >> 16);
}

// A-operand frag (16x32 bf16): lane&15 = M row; lanes<16: K {k..k+15},
// lanes>=16: K {k+16..k+31}
DEV bf16x16 frag_a(const unsigned short* row, int k0, int lane) {
  int half = (lane >> 4) & 1;
  FragU f;
  f.q[0] = ld16(row + k0 + 16 * half);
  f.q[1] = ld16(row + k0 + 16 * half + 8);
  return f.v;
}
// A frag with upper 16 K-values zero (for K=48 head dim padded to 64)
DEV bf16x16 frag_a_lo(const unsigned short* row, int k0, int lane) {
  int half = (lane >> 4) & 1;
  FragU f;
  f.q[0] = half ? make_uint4(0, 0, 0, 0) : ld16(row + k0);
  f.q[1] = half ? make_uint4(0, 0, 0, 0) : ld16(row + k0 + 8);
  return f.v;
}
// B-operand frag (32x16 bf16): lane&15 = N col; lanes<16: K {k..k+15},
// lanes>=16: K {k+16..k+31}. colPtr = row of W (== column of B).
DEV bf16x16 frag_b(const unsigned short* colPtr, int k0, int lane) {
  int half = (lane >> 4) & 1;
  int kk = k0 + 16 * half;
  FragU f;
  f.q[0] = ld16(colPtr + kk);
  f.q[1] = ld16(colPtr + kk + 8);
  return f.v;
}
DEV bf16x16 frag_b_guard(const unsigned short* colPtr, int k0, int lane, int kmax) {
  int half = (lane >> 4) & 1;
  int kk = k0 + 16 * half;
  FragU f;
  if (kk + 16 <= kmax) { f.q[0] = ld16(colPtr + kk); f.q[1] = ld16(colPtr + kk + 8); }
  else                 { f.q[0] = make_uint4(0,0,0,0); f.q[1] = make_uint4(0,0,0,0); }
  return f.v;
}

DEV f32x8 wmma_bf16(bf16x16 a, bf16x16 b, f32x8 c) {
  return __builtin_amdgcn_wmma_f32_16x16x32_bf16(false, a, false, b, (short)0, c,
                                                 false, false);
}

// positionalencoding2d value at (channel c, h, w) for given d_model
DEV float pe2(int dmodel, int c, int h, int w) {
  int dh = dmodel >> 1;
  int cc  = (c < dh) ? c : (c - dh);
  int pos = (c < dh) ? w : h;
  int k = cc >> 1;
  float div = __expf(-(float)(2 * k) * 9.210340371976184f / (float)dh); // ln(10000)
  float a = (float)pos * div;
  return (cc & 1) ? __cosf(a) : __sinf(a);
}

// ---------------------------------------------------------------- prep kernels
__global__ __launch_bounds__(256) void cvt_bf16_k(const float* __restrict__ src,
                                                  unsigned short* __restrict__ dst,
                                                  int n) {
  int i = blockIdx.x * blockDim.x + threadIdx.x;
  if (i < n) dst[i] = f2bf(src[i]);
}

// up_w (O=384, I=384, 3,3) -> wtap[tap][o][c] bf16 (contiguous K for B frags)
__global__ __launch_bounds__(256) void upw_k(const float* __restrict__ up,
                                             unsigned short* __restrict__ wtap) {
  int i = blockIdx.x * blockDim.x + threadIdx.x;
  const int n = 9 * D2c * D2c;
  if (i >= n) return;
  int tap = i / (D2c * D2c);
  int r = i % (D2c * D2c);
  int o = r / D2c, c = r % D2c;
  wtap[i] = f2bf(up[(size_t)(o * D2c + c) * 9 + tap]);
}

// y+PE -> padded bf16 token buffer (halo = 0) + unpadded bf16 token buffer;
// s+PE -> bf16 token buffer + fp32 copy (for the sigmoid gate).
__global__ __launch_bounds__(256) void prep_k(const float* __restrict__ y,
                                              const float* __restrict__ s,
                                              unsigned short* __restrict__ ypePad,
                                              unsigned short* __restrict__ ypeT,
                                              unsigned short* __restrict__ speT,
                                              float* __restrict__ speF) {
  int p = blockIdx.x * blockDim.x + threadIdx.x;
  if (p >= BSc * PTOK) return;
  int b = p / PTOK, r = p % PTOK;
  int ph = r / PW, pw = r % PW;
  unsigned short* dstPad = ypePad + (size_t)p * D2c;
  if (ph == 0 || ph == PW - 1 || pw == 0 || pw == PW - 1) {
    for (int c = 0; c < D2c; c++) dstPad[c] = 0;
    return;
  }
  int h = ph - 1, w = pw - 1;
  int hw = h * Wc + w;
  int t = b * NYc + hw;
  for (int c = 0; c < D2c; c++) {
    float v = y[((size_t)(b * D2c + c)) * NYc + hw] + pe2(D2c, c, h, w);
    unsigned short bv = f2bf(v);
    dstPad[c] = bv;
    ypeT[(size_t)t * D2c + c] = bv;
  }
  for (int c = 0; c < Dc; c++) {
    float v = s[((size_t)(b * Dc + c)) * NYc + hw] + pe2(Dc, c, h, w);
    speT[(size_t)t * Dc + c] = f2bf(v);
    speF[(size_t)t * Dc + c] = v;
  }
}

// ---------------------------------------------------------------- WMMA GEMM
// C[t, n] = act(bn(A[t, :] . Wm[n, :] + bias[n]))
// wave tile: 16 tokens x 64 outs (4 n-tiles); block: 4 waves along tokens.
// MODE 0: +bias           -> bf16 outB
// MODE 1: +bias, bn, relu -> bf16 outB
// MODE 2: +bias, bn, relu -> d_out NCHW at channel offset Dc (y_c3 half)
// MODE 3: +bias, bn, sigmoid, * (s+PE) -> d_out NCHW at channel offset 0
template <int MODE>
__global__ __launch_bounds__(128) void gemm_k(const unsigned short* __restrict__ A,
                                              const unsigned short* __restrict__ Wm,
                                              const float* __restrict__ bias,
                                              const float* __restrict__ bng,
                                              const float* __restrict__ bnb,
                                              unsigned short* __restrict__ outB,
                                              float* __restrict__ outF,
                                              const float* __restrict__ sAux,
                                              int N, int K) {
  const int lane = threadIdx.x & 31;
  const int wave = threadIdx.x >> 5;
  const int m0 = (blockIdx.x * 4 + wave) * 16;
  const int n0 = blockIdx.y * 64;
  const int col = lane & 15, half = (lane >> 4) & 1;
  if (m0 >= NTOK) return;

  const unsigned short* aRow = A + (size_t)(m0 + col) * K;
  f32x8 acc[4];
#pragma unroll
  for (int i = 0; i < 4; i++) acc[i] = (f32x8){0, 0, 0, 0, 0, 0, 0, 0};

  for (int k = 0; k < K; k += 32) {
    bf16x16 a = frag_a(aRow, k, lane);
#pragma unroll
    for (int i = 0; i < 4; i++) {
      bf16x16 b = frag_b(Wm + (size_t)(n0 + i * 16 + col) * K, k, lane);
      acc[i] = wmma_bf16(a, b, acc[i]);
    }
  }

#pragma unroll
  for (int i = 0; i < 4; i++) {
    int n = n0 + i * 16 + col;
    float bia = bias[n];
    float g = 1.f, bb = 0.f;
    if (MODE == 1 || MODE == 2 || MODE == 3) { g = bng[n]; bb = bnb[n]; }
#pragma unroll
    for (int rr = 0; rr < 8; rr++) {
      int m = m0 + rr + 8 * half;
      float v = acc[i][rr] + bia;
      if (MODE == 1 || MODE == 2 || MODE == 3) v = v * g + bb;
      if (MODE == 1 || MODE == 2) v = fmaxf(v, 0.f);
      if (MODE == 0 || MODE == 1) {
        outB[(size_t)m * N + n] = f2bf(v);
      } else {
        int bi = m / NYc, hw = m % NYc;
        if (MODE == 2) {
          outF[((size_t)(bi * D2c) + Dc + n) * NYc + hw] = v;
        } else {  // MODE 3: sigmoid gate * (s + PE)
          float gg = 1.f / (1.f + __expf(-v));
          outF[((size_t)(bi * D2c) + n) * NYc + hw] = gg * sAux[(size_t)m * Dc + n];
        }
      }
    }
  }
}

// ----------------------------------------------------- conv3x3 implicit GEMM
__global__ __launch_bounds__(128) void conv3_k(const unsigned short* __restrict__ ypePad,
                                               const unsigned short* __restrict__ wtap,
                                               const float* __restrict__ bias,
                                               unsigned short* __restrict__ outB) {
  const int lane = threadIdx.x & 31;
  const int wave = threadIdx.x >> 5;
  const int m0 = (blockIdx.x * 4 + wave) * 16;
  const int n0 = blockIdx.y * 64;
  const int col = lane & 15, half = (lane >> 4) & 1;

  int t = m0 + col;
  int b = t / NYc, hw = t % NYc;
  int h = hw / Wc, w = hw % Wc;
  const unsigned short* base =
      ypePad + (size_t)(b * PTOK + (h + 1) * PW + (w + 1)) * D2c;

  f32x8 acc[4];
#pragma unroll
  for (int i = 0; i < 4; i++) acc[i] = (f32x8){0, 0, 0, 0, 0, 0, 0, 0};

#pragma unroll
  for (int tap = 0; tap < 9; ++tap) {
    int dy = tap / 3 - 1, dx = tap % 3 - 1;
    const unsigned short* aRow = base + (ptrdiff_t)(dy * PW + dx) * D2c;
    const unsigned short* wBase = wtap + (size_t)tap * D2c * D2c;
    for (int k = 0; k < D2c; k += 32) {
      bf16x16 a = frag_a(aRow, k, lane);
#pragma unroll
      for (int i = 0; i < 4; i++) {
        bf16x16 bf = frag_b(wBase + (size_t)(n0 + i * 16 + col) * D2c, k, lane);
        acc[i] = wmma_bf16(a, bf, acc[i]);
      }
    }
  }
#pragma unroll
  for (int i = 0; i < 4; i++) {
    int n = n0 + i * 16 + col;
    float bia = bias[n];
#pragma unroll
    for (int rr = 0; rr < 8; rr++) {
      int m = m0 + rr + 8 * half;
      outB[(size_t)m * D2c + n] = f2bf(acc[i][rr] + bia);
    }
  }
}

// ------------------------------------------------------------ flash attention
// One block = (batch b, head, 64-query block); 4 waves x 16-query tiles.
// Loops over 32-key chunks; K/V chunks staged in LDS; probabilities repacked
// C-layout -> A-layout through LDS for the P@V WMMA.
__global__ __launch_bounds__(128) void flash_k(const unsigned short* __restrict__ Qb,
                                               const unsigned short* __restrict__ Kb,
                                               const unsigned short* __restrict__ Vb,
                                               unsigned short* __restrict__ Zb) {
  __shared__ __align__(16) unsigned short Kc[32 * HDQK];  // [key][d]  32x48
  __shared__ __align__(16) unsigned short Vt[HDV * 32];   // [dim][key] 24x32
  __shared__ __align__(16) unsigned short Pl[4][16 * 32]; // per-wave P tile

  const int lane = threadIdx.x & 31;
  const int wave = threadIdx.x >> 5;
  const int tid = threadIdx.x;
  const int col = lane & 15, half = (lane >> 4) & 1;

  int blk = blockIdx.x;
  int qblk = blk % (NYc / 64);
  int head = (blk / (NYc / 64)) % NHc;
  int b = blk / ((NYc / 64) * NHc);
  int q0 = qblk * 64 + wave * 16;

  const unsigned short* qRow = Qb + (size_t)(b * NYc + q0 + col) * D2c + head * HDQK;
  bf16x16 qa0 = frag_a(qRow, 0, lane);       // head-dim 0..31
  bf16x16 qa1 = frag_a_lo(qRow, 32, lane);   // head-dim 32..47 (+16 zero pad)

  float mrow[8], lrow[8];
  f32x8 o0 = (f32x8){0, 0, 0, 0, 0, 0, 0, 0};
  f32x8 o1 = (f32x8){0, 0, 0, 0, 0, 0, 0, 0};
#pragma unroll
  for (int r = 0; r < 8; r++) { mrow[r] = -1e30f; lrow[r] = 0.f; }
  const float sc = 0.07216878364870323f;  // 1/sqrt(192)

  for (int j = 0; j < NYc; j += 32) {
    __syncthreads();  // protect previous-iteration LDS reads
    for (int idx = tid; idx < 32 * HDQK; idx += 128) {
      int kk = idx / HDQK, d = idx % HDQK;
      Kc[idx] = Kb[(size_t)(b * NYc + j + kk) * D2c + head * HDQK + d];
    }
    for (int idx = tid; idx < HDV * 32; idx += 128) {
      int n = idx / 32, kk = idx % 32;
      Vt[idx] = Vb[(size_t)(b * NYc + j + kk) * Dc + head * HDV + n];
    }
    __syncthreads();

    f32x8 s0 = (f32x8){0, 0, 0, 0, 0, 0, 0, 0};
    f32x8 s1 = (f32x8){0, 0, 0, 0, 0, 0, 0, 0};
    s0 = wmma_bf16(qa0, frag_b(&Kc[col * HDQK], 0, lane), s0);
    s0 = wmma_bf16(qa1, frag_b_guard(&Kc[col * HDQK], 32, lane, HDQK), s0);
    s1 = wmma_bf16(qa0, frag_b(&Kc[(16 + col) * HDQK], 0, lane), s1);
    s1 = wmma_bf16(qa1, frag_b_guard(&Kc[(16 + col) * HDQK], 32, lane, HDQK), s1);

#pragma unroll
    for (int r = 0; r < 8; r++) {
      float a0 = s0[r] * sc, a1 = s1[r] * sc;
      float mx = fmaxf(a0, a1);
      mx = fmaxf(mx, __shfl_xor(mx, 1, 32));
      mx = fmaxf(mx, __shfl_xor(mx, 2, 32));
      mx = fmaxf(mx, __shfl_xor(mx, 4, 32));
      mx = fmaxf(mx, __shfl_xor(mx, 8, 32));
      float mnew = fmaxf(mrow[r], mx);
      float alpha = __expf(mrow[r] - mnew);
      float p0 = __expf(a0 - mnew), p1 = __expf(a1 - mnew);
      float ps = p0 + p1;
      ps += __shfl_xor(ps, 1, 32);
      ps += __shfl_xor(ps, 2, 32);
      ps += __shfl_xor(ps, 4, 32);
      ps += __shfl_xor(ps, 8, 32);
      lrow[r] = lrow[r] * alpha + ps;
      mrow[r] = mnew;
      o0[r] *= alpha;
      o1[r] *= alpha;
      Pl[wave][(r + 8 * half) * 32 + col] = f2bf(p0);
      Pl[wave][(r + 8 * half) * 32 + 16 + col] = f2bf(p1);
    }
    __syncthreads();

    bf16x16 pa = frag_a(&Pl[wave][col * 32], 0, lane);
    o0 = wmma_bf16(pa, frag_b(&Vt[col * 32], 0, lane), o0);
    bf16x16 v1;
    {
      FragU f;
      if (col < HDV - 16) {  // dims 16..23, pad cols 24..31 with zeros
        const unsigned short* vp = &Vt[(16 + col) * 32];
        int kk = 16 * half;
        f.q[0] = ld16(vp + kk);
        f.q[1] = ld16(vp + kk + 8);
      } else {
        f.q[0] = make_uint4(0, 0, 0, 0);
        f.q[1] = make_uint4(0, 0, 0, 0);
      }
      v1 = f.v;
    }
    o1 = wmma_bf16(pa, v1, o1);
  }

#pragma unroll
  for (int r = 0; r < 8; r++) {
    float inv = 1.f / lrow[r];
    int q = q0 + r + 8 * half;
    size_t t = (size_t)(b * NYc + q);
    Zb[t * Dc + head * HDV + col] = f2bf(o0[r] * inv);
    if (col < HDV - 16) Zb[t * Dc + head * HDV + 16 + col] = f2bf(o1[r] * inv);
  }
}

// ------------------------------------------------------------------- launcher
extern "C" void kernel_launch(void* const* d_in, const int* in_sizes, int n_in,
                              void* d_out, int out_size, void* d_ws, size_t ws_size,
                              hipStream_t stream) {
  (void)in_sizes; (void)n_in; (void)out_size; (void)ws_size;
  const float* y       = (const float*)d_in[0];
  const float* s       = (const float*)d_in[1];
  const float* wq_w    = (const float*)d_in[2];
  const float* wq_b    = (const float*)d_in[3];
  const float* wk_w    = (const float*)d_in[4];
  const float* wk_b    = (const float*)d_in[5];
  const float* wv_w    = (const float*)d_in[6];
  const float* wv_b    = (const float*)d_in[7];
  const float* conv1_w = (const float*)d_in[8];
  const float* conv1_b = (const float*)d_in[9];
  const float* bn1_g   = (const float*)d_in[10];
  const float* bn1_b   = (const float*)d_in[11];
  const float* conv2_w = (const float*)d_in[12];
  const float* conv2_b = (const float*)d_in[13];
  const float* bn2_g   = (const float*)d_in[14];
  const float* bn2_b   = (const float*)d_in[15];
  const float* up_w    = (const float*)d_in[16];
  const float* up_b    = (const float*)d_in[17];
  const float* conv3_w = (const float*)d_in[18];
  const float* conv3_b = (const float*)d_in[19];
  const float* bn3_g   = (const float*)d_in[20];
  const float* bn3_b   = (const float*)d_in[21];
  const float* sig_w   = (const float*)d_in[22];
  const float* sig_b   = (const float*)d_in[23];
  const float* bnsig_g = (const float*)d_in[24];
  const float* bnsig_b = (const float*)d_in[25];
  float* out = (float*)d_out;

  uint8_t* base = (uint8_t*)d_ws;
  size_t off = 0;
  auto alloc = [&](size_t elems, size_t elsz) -> void* {
    void* r = base + off;
    off += (elems * elsz + 255) & ~(size_t)255;
    return r;
  };
  unsigned short* ypePad = (unsigned short*)alloc((size_t)BSc * PTOK * D2c, 2);
  unsigned short* ypeT   = (unsigned short*)alloc((size_t)NTOK * D2c, 2);
  unsigned short* speT   = (unsigned short*)alloc((size_t)NTOK * Dc, 2);
  float*          speF   = (float*)alloc((size_t)NTOK * Dc, 4);
  unsigned short* yc1    = (unsigned short*)alloc((size_t)NTOK * D2c, 2);
  unsigned short* sc2    = (unsigned short*)alloc((size_t)NTOK * Dc, 2);
  unsigned short* Qb     = (unsigned short*)alloc((size_t)NTOK * D2c, 2);
  unsigned short* Kb     = (unsigned short*)alloc((size_t)NTOK * D2c, 2);
  unsigned short* Vb     = (unsigned short*)alloc((size_t)NTOK * Dc, 2);
  unsigned short* yc3a   = (unsigned short*)alloc((size_t)NTOK * D2c, 2);
  unsigned short* Zb     = (unsigned short*)alloc((size_t)NTOK * Dc, 2);
  unsigned short* w1b    = (unsigned short*)alloc((size_t)D2c * D2c, 2);
  unsigned short* wqb    = (unsigned short*)alloc((size_t)D2c * D2c, 2);
  unsigned short* wkb    = (unsigned short*)alloc((size_t)D2c * D2c, 2);
  unsigned short* w2b    = (unsigned short*)alloc((size_t)Dc * Dc, 2);
  unsigned short* wvb    = (unsigned short*)alloc((size_t)Dc * Dc, 2);
  unsigned short* w3b    = (unsigned short*)alloc((size_t)Dc * D2c, 2);
  unsigned short* wsb    = (unsigned short*)alloc((size_t)Dc * Dc, 2);
  unsigned short* wtap   = (unsigned short*)alloc((size_t)9 * D2c * D2c, 2);

  auto cdiv = [](int a, int b) { return (a + b - 1) / b; };

  // weight conversions
  cvt_bf16_k<<<cdiv(D2c * D2c, 256), 256, 0, stream>>>(conv1_w, w1b, D2c * D2c);
  cvt_bf16_k<<<cdiv(D2c * D2c, 256), 256, 0, stream>>>(wq_w, wqb, D2c * D2c);
  cvt_bf16_k<<<cdiv(D2c * D2c, 256), 256, 0, stream>>>(wk_w, wkb, D2c * D2c);
  cvt_bf16_k<<<cdiv(Dc * Dc, 256), 256, 0, stream>>>(conv2_w, w2b, Dc * Dc);
  cvt_bf16_k<<<cdiv(Dc * Dc, 256), 256, 0, stream>>>(wv_w, wvb, Dc * Dc);
  cvt_bf16_k<<<cdiv(Dc * D2c, 256), 256, 0, stream>>>(conv3_w, w3b, Dc * D2c);
  cvt_bf16_k<<<cdiv(Dc * Dc, 256), 256, 0, stream>>>(sig_w, wsb, Dc * Dc);
  upw_k<<<cdiv(9 * D2c * D2c, 256), 256, 0, stream>>>(up_w, wtap);

  // tokens + PE
  prep_k<<<cdiv(BSc * PTOK, 256), 256, 0, stream>>>(y, s, ypePad, ypeT, speT, speF);

  const dim3 blk128(128);
  const dim3 gY(NTOK / 64, D2c / 64);  // (72, 6)
  const dim3 gS(NTOK / 64, Dc / 64);   // (72, 3)

  // y_c1 = relu(bn1(ype @ conv1^T))
  gemm_k<1><<<gY, blk128, 0, stream>>>(ypeT, w1b, conv1_b, bn1_g, bn1_b, yc1,
                                       nullptr, nullptr, D2c, D2c);
  // s_c2 = relu(bn2(spe @ conv2^T))
  gemm_k<1><<<gS, blk128, 0, stream>>>(speT, w2b, conv2_b, bn2_g, bn2_b, sc2,
                                       nullptr, nullptr, Dc, Dc);
  // Q, K, V projections
  gemm_k<0><<<gY, blk128, 0, stream>>>(yc1, wqb, wq_b, nullptr, nullptr, Qb,
                                       nullptr, nullptr, D2c, D2c);
  gemm_k<0><<<gY, blk128, 0, stream>>>(yc1, wkb, wk_b, nullptr, nullptr, Kb,
                                       nullptr, nullptr, D2c, D2c);
  gemm_k<0><<<gS, blk128, 0, stream>>>(sc2, wvb, wv_b, nullptr, nullptr, Vb,
                                       nullptr, nullptr, Dc, Dc);
  // 'up' branch: conv3x3 then conv3 1x1 (+bn3, relu) -> d_out[:, 192:384]
  conv3_k<<<gY, blk128, 0, stream>>>(ypePad, wtap, up_b, yc3a);
  gemm_k<2><<<gS, blk128, 0, stream>>>(yc3a, w3b, conv3_b, bn3_g, bn3_b, nullptr,
                                       out, nullptr, Dc, D2c);
  // attention -> Z tokens
  flash_k<<<BSc * NHc * (NYc / 64), blk128, 0, stream>>>(Qb, Kb, Vb, Zb);
  // sigmoid gate * (s+PE) -> d_out[:, 0:192]
  gemm_k<3><<<gS, blk128, 0, stream>>>(Zb, wsb, sig_b, bnsig_g, bnsig_b, nullptr,
                                       out, speF, Dc, Dc);
}
